// AttentionLayer_26328149525073
// MI455X (gfx1250) — matrix operand order repaired and checked by reading, not compiled, not run on hardware
//
#include <hip/hip_runtime.h>
#include <stdint.h>

#define B_  8
#define T_  2048
#define NX_ 512
#define NH_ 512

typedef __attribute__((ext_vector_type(16))) __bf16 v16bf;
typedef __attribute__((ext_vector_type(2)))  __bf16 v2bf;
typedef __attribute__((ext_vector_type(8)))  float  v8f;
typedef __attribute__((ext_vector_type(2)))  float  v2f;
typedef __attribute__((ext_vector_type(4)))  unsigned int v4u;
typedef __attribute__((ext_vector_type(8)))  int    v8i;
typedef __attribute__((ext_vector_type(4)))  int    v4i;

union AF { v16bf v; uint32_t u[8]; };
union CF { v8f   v; float    f[8]; };

#if defined(__has_builtin)
#if __has_builtin(__builtin_amdgcn_tensor_load_to_lds) && __has_builtin(__builtin_amdgcn_s_wait_tensorcnt)
#define HAVE_TDM 1
#endif
#endif
#ifndef HAVE_TDM
#define HAVE_TDM 0
#endif

// Native bf16 converts: scalar cast -> v_cvt_bf16-class op; vector convert ->
// packed v_cvt_pk_bf16_f32 where the backend supports it.  (RNE by default.)
__device__ __forceinline__ uint16_t f2bf(float f) {
  union { __bf16 h; uint16_t u; } cv;
  cv.h = (__bf16)f;
  return cv.u;
}
__device__ __forceinline__ uint32_t pack2bf(float lo, float hi) {
  union { v2bf h; uint32_t u; } cv;
  v2f in;
  in[0] = lo;
  in[1] = hi;
  cv.h = __builtin_convertvector(in, v2bf);
  return cv.u;
}

__device__ __forceinline__ v8f wmma_bf16(const AF& a, const AF& b, v8f c) {
  return __builtin_amdgcn_wmma_f32_16x16x32_bf16(false, a.v, false, b.v,
                                                 (short)0, c, false, false);
}

// A-frag (16x32 bf16): lanes0-15 M=0..15; VGPR v holds K pair k = 16*(v>>2)+8*half+2*(v&3).
__device__ __forceinline__ void load_afrag(AF& a, const uint16_t* base,
                                           int rowbase, int ld, int kbase, int lane) {
  const int m = lane & 15, half = lane >> 4;
  const uint16_t* p = base + (size_t)(rowbase + m) * ld + kbase;
#pragma unroll
  for (int v = 0; v < 8; ++v) {
    const int k = ((v >> 2) << 4) + (half << 3) + ((v & 3) << 1);
    a.u[v] = *(const uint32_t*)(p + k);
  }
}
// B-frag (32x16 bf16) from a K-contiguous tile: element (k,n) at base[(colbase+n)*ld + kbase + k].
__device__ __forceinline__ void load_bfrag(AF& b, const uint16_t* base,
                                           int colbase, int ld, int kbase, int lane) {
  const int n = lane & 15, half = lane >> 4;
  const uint16_t* p = base + (size_t)(colbase + n) * ld + kbase;
#pragma unroll
  for (int v = 0; v < 8; ++v) b.u[v] = *(const uint32_t*)(p + ((v >> 2) << 4) + (half << 3) + ((v & 3) << 1));
}

#if HAVE_TDM
// ---- Tensor Data Mover: 2D bf16 tile (dim0 contiguous) global -> LDS -------
__device__ __forceinline__ void tdm_issue(v4u g0, v8i g1) {
  v4i z4 = {0, 0, 0, 0};
#if __clang_major__ >= 23
  v8i z8 = {0, 0, 0, 0, 0, 0, 0, 0};
  __builtin_amdgcn_tensor_load_to_lds(g0, g1, z4, z4, z8, 0);
#else
  __builtin_amdgcn_tensor_load_to_lds(g0, g1, z4, z4, 0);
#endif
}
__device__ __forceinline__ void tdm_load_2d(const void* lds_dst, const void* gsrc,
                                            uint32_t tile0, uint32_t tile1,
                                            uint32_t dim0, uint32_t dim1,
                                            uint64_t stride0) {
  const uint64_t ga = (uint64_t)(uintptr_t)gsrc;
  v4u g0;
  g0[0] = 1u;                                        // count=1 (user descriptor)
  g0[1] = (uint32_t)(uintptr_t)lds_dst;              // lds_addr (bytes)
  g0[2] = (uint32_t)ga;                              // global_addr[31:0]
  g0[3] = (uint32_t)((ga >> 32) & 0x01FFFFFFu) | (2u << 30);  // addr[56:32] | type=2
  v8i g1;
  g1[0] = (int)(1u << 16);                           // data_size=1 (2B); mask=0
  g1[1] = (int)((dim0 & 0xFFFFu) << 16);             // tensor_dim0[15:0] @ bit48
  g1[2] = (int)((dim0 >> 16) | ((dim1 & 0xFFFFu) << 16));
  g1[3] = (int)((dim1 >> 16) | (tile0 << 16));       // tile_dim0 @ [127:112]
  g1[4] = (int)(tile1 & 0xFFFFu);                    // tile_dim1; tile_dim2=0
  g1[5] = (int)(uint32_t)(stride0 & 0xFFFFFFFFu);    // tensor_dim0_stride[31:0]
  g1[6] = (int)(uint32_t)((stride0 >> 32) & 0xFFFFu);
  g1[7] = 0;
  tdm_issue(g0, g1);
}
#endif

// ---------------------------------------------------------------------------
// Kernel 1: fused QKV projection.  x[16384,512] fp32 @ W[512,512] fp32 -> bf16.
// Q,K stored row-major [token][feat]; V stored TRANSPOSED [b][feat][T] so the
// flash kernel's P@V B-fragments are K-contiguous without an LDS transpose.
// ---------------------------------------------------------------------------
__global__ __launch_bounds__(256) void qkv_gemm(
    const float* __restrict__ x,
    const float* __restrict__ Wq, const float* __restrict__ Wk,
    const float* __restrict__ Wv,
    uint16_t* __restrict__ Qo, uint16_t* __restrict__ Ko, uint16_t* __restrict__ Vo)
{
  __shared__ uint16_t As[128 * 32];   // [row][k]
  __shared__ uint16_t Bs[128 * 32];   // transposed: [n][k]

  const float* W = (blockIdx.z == 0) ? Wq : (blockIdx.z == 1 ? Wk : Wv);

  const int tid = threadIdx.x, lane = tid & 31, wid = tid >> 5;
  const int row0 = blockIdx.x * 128, n0 = blockIdx.y * 128;
  const int wrow = (wid >> 2) * 64, wcol = (wid & 3) * 32;

  CF acc[4][2];
#pragma unroll
  for (int mt = 0; mt < 4; ++mt)
#pragma unroll
    for (int nt = 0; nt < 2; ++nt)
#pragma unroll
      for (int r = 0; r < 8; ++r) acc[mt][nt].f[r] = 0.f;

  for (int k0 = 0; k0 < NX_; k0 += 32) {
    uint32_t* AsU = (uint32_t*)As;
#pragma unroll
    for (int j = 0; j < 4; ++j) {            // 128x32 fp32 -> bf16
      const int e = tid + j * 256;
      const int r = e >> 3, kc = (e & 7) << 2;
      const float4 f = *(const float4*)(x + (size_t)(row0 + r) * NX_ + k0 + kc);
      AsU[r * 16 + (kc >> 1)]     = pack2bf(f.x, f.y);
      AsU[r * 16 + (kc >> 1) + 1] = pack2bf(f.z, f.w);
    }
#pragma unroll
    for (int j = 0; j < 4; ++j) {            // 32x128 fp32 -> bf16, transposed
      const int e = tid + j * 256;
      const int kr = e >> 5, nc = (e & 31) << 2;
      const float4 f = *(const float4*)(W + (size_t)(k0 + kr) * NH_ + n0 + nc);
      Bs[(nc + 0) * 32 + kr] = f2bf(f.x);
      Bs[(nc + 1) * 32 + kr] = f2bf(f.y);
      Bs[(nc + 2) * 32 + kr] = f2bf(f.z);
      Bs[(nc + 3) * 32 + kr] = f2bf(f.w);
    }
    __syncthreads();

    AF a[4], b[2];
#pragma unroll
    for (int mt = 0; mt < 4; ++mt) load_afrag(a[mt], As, wrow + mt * 16, 32, 0, lane);
#pragma unroll
    for (int nt = 0; nt < 2; ++nt) load_bfrag(b[nt], Bs, wcol + nt * 16, 32, 0, lane);
#pragma unroll
    for (int mt = 0; mt < 4; ++mt)
#pragma unroll
      for (int nt = 0; nt < 2; ++nt)
        acc[mt][nt].v = wmma_bf16(a[mt], b[nt], acc[mt][nt].v);
    __syncthreads();
  }

  const int half = lane >> 4, nn = lane & 15;
  if (blockIdx.z == 2) {
    // V: transposed store.  C-frag lane owns 8 consecutive rows t at fixed col
    // -> one 16B packed store per tile into V^T[b][feat][T].
    const int bb = row0 >> 11;                 // batch (128 | 2048)
    const int tloc0 = (row0 & (T_ - 1)) + wrow + (half << 3);
#pragma unroll
    for (int mt = 0; mt < 4; ++mt)
#pragma unroll
      for (int nt = 0; nt < 2; ++nt) {
        const int gn = n0 + wcol + nt * 16 + nn;
        uint4 q;
        q.x = pack2bf(acc[mt][nt].f[0], acc[mt][nt].f[1]);
        q.y = pack2bf(acc[mt][nt].f[2], acc[mt][nt].f[3]);
        q.z = pack2bf(acc[mt][nt].f[4], acc[mt][nt].f[5]);
        q.w = pack2bf(acc[mt][nt].f[6], acc[mt][nt].f[7]);
        *(uint4*)(Vo + ((size_t)bb * NH_ + gn) * T_ + tloc0 + mt * 16) = q;
      }
  } else {
    uint16_t* Out = (blockIdx.z == 0) ? Qo : Ko;
#pragma unroll
    for (int mt = 0; mt < 4; ++mt)
#pragma unroll
      for (int nt = 0; nt < 2; ++nt)
#pragma unroll
        for (int r = 0; r < 8; ++r) {
          const int gm = row0 + wrow + mt * 16 + r + (half << 3);
          const int gn = n0 + wcol + nt * 16 + nn;
          Out[(size_t)gm * NH_ + gn] = f2bf(acc[mt][nt].f[r]);
        }
  }
}

// ---------------------------------------------------------------------------
// Kernel 2: flash attention.  Block = 128 queries (8 waves x 16).  K/V tiles
// (32 keys) double-buffered in LDS via the Tensor Data Mover; wave 0 issues
// the next block's DMA while all waves compute on the current one.
// ---------------------------------------------------------------------------
__global__ __launch_bounds__(256) void flash_attn(
    const uint16_t* __restrict__ Q, const uint16_t* __restrict__ K,
    const uint16_t* __restrict__ Vt, const int* __restrict__ mask,
    uint16_t* __restrict__ Aout)
{
  extern __shared__ uint16_t smem[];
  // layout: [buf0: Ks 32x512 | Vt 512x32][buf1: same][Ps 8x16x32]
  uint16_t* Ps = smem + 4 * 16384;

  const int tid = threadIdx.x, lane = tid & 31, wid = tid >> 5;
  const int half = lane >> 4, nn = lane & 15;
  const int b  = blockIdx.y;
  const int q0 = blockIdx.x * 128 + wid * 16;

  // Wave-resident 16x512 Q tile as 16 A-fragments.
  const uint32_t* Qu = (const uint32_t*)(Q + ((size_t)b * T_ + q0) * NH_);
  AF qf[16];
#pragma unroll
  for (int c = 0; c < 16; ++c)
#pragma unroll
    for (int v = 0; v < 8; ++v) {
      const int k = c * 32 + ((v >> 2) << 4) + (half << 3) + ((v & 3) << 1);
      qf[c].u[v] = Qu[((size_t)nn * NH_ + k) >> 1];
    }

  CF o[32];
#pragma unroll
  for (int ch = 0; ch < 32; ++ch)
#pragma unroll
    for (int r = 0; r < 8; ++r) o[ch].f[r] = 0.f;

  float rm[8], rl[8];
#pragma unroll
  for (int r = 0; r < 8; ++r) { rm[r] = -3.0e38f; rl[r] = 0.f; }

  const float scale = 0.022097086912079608f;  // 1/sqrt(T) per the reference
  uint16_t*  Pw   = Ps + wid * (16 * 32);
  const int* mrow = mask + (size_t)b * T_;

#if HAVE_TDM
  if (wid == 0) {   // prologue: DMA block 0 into buffer 0
    tdm_load_2d(smem,         K + ((size_t)b * T_) * NH_, NX_, 32, NX_, T_, NX_);
    tdm_load_2d(smem + 16384, Vt + (size_t)b * NH_ * T_,  32, NH_, T_, NH_, T_);
  }
#endif

  for (int kb = 0, it = 0; kb < T_; kb += 32, ++it) {
#if HAVE_TDM
    const int cur = it & 1, nxt = cur ^ 1;
    uint16_t* Ks = smem + cur * (2 * 16384);
    uint16_t* Vs = Ks + 16384;
    if (wid == 0) {
      if (kb + 32 < T_) {
        uint16_t* Kn = smem + nxt * (2 * 16384);
        tdm_load_2d(Kn,         K + ((size_t)b * T_ + kb + 32) * NH_,
                    NX_, 32, NX_, T_, NX_);
        tdm_load_2d(Kn + 16384, Vt + (size_t)b * NH_ * T_ + kb + 32,
                    32, NH_, T_, NH_, T_);
        __builtin_amdgcn_s_wait_tensorcnt(2);   // current block's pair done
      } else {
        __builtin_amdgcn_s_wait_tensorcnt(0);
      }
    }
    __syncthreads();
#else
    uint16_t* Ks = smem;
    uint16_t* Vs = smem + 16384;
    {
      const uint32_t* Kg = (const uint32_t*)(K + ((size_t)b * T_ + kb) * NH_);
      uint32_t* KsU = (uint32_t*)Ks;
      uint32_t* VsU = (uint32_t*)Vs;
#pragma unroll
      for (int j = 0; j < 32; ++j) {
        const int e = tid + j * 256;
        KsU[e] = Kg[e];
      }
#pragma unroll
      for (int j = 0; j < 32; ++j) {           // straight copy of V^T rows
        const int e = tid + j * 256;
        const int f = e >> 4, kd = e & 15;
        VsU[e] = *(const uint32_t*)(Vt + ((size_t)b * NH_ + f) * T_ + kb + kd * 2);
      }
    }
    __syncthreads();
#endif

    // ---- S = Q Kt for this wave's 16 queries x 32 keys ----
    CF s0, s1;
#pragma unroll
    for (int r = 0; r < 8; ++r) { s0.f[r] = 0.f; s1.f[r] = 0.f; }
#pragma unroll
    for (int c = 0; c < 16; ++c) {
      AF b0, b1;
      load_bfrag(b0, Ks, 0,  NH_, c * 32, lane);
      load_bfrag(b1, Ks, 16, NH_, c * 32, lane);
      s0.v = wmma_bf16(qf[c], b0, s0.v);
      s1.v = wmma_bf16(qf[c], b1, s1.v);
    }

    // ---- mask + online softmax (row = r + 8*half, col = nn) ----
    const int mk0 = mrow[kb + nn];
    const int mk1 = mrow[kb + 16 + nn];
    float al[8];
#pragma unroll
    for (int r = 0; r < 8; ++r) {
      float a0 = mk0 ? s0.f[r] * scale : -3.0e38f;
      float a1 = mk1 ? s1.f[r] * scale : -3.0e38f;
      float t = fmaxf(a0, a1);
      t = fmaxf(t, __shfl_xor(t, 1, 32));
      t = fmaxf(t, __shfl_xor(t, 2, 32));
      t = fmaxf(t, __shfl_xor(t, 4, 32));
      t = fmaxf(t, __shfl_xor(t, 8, 32));
      const float mnew = fmaxf(rm[r], t);
      al[r] = __expf(rm[r] - mnew);
      rm[r] = mnew;
      const float p0 = __expf(a0 - mnew);
      const float p1 = __expf(a1 - mnew);
      float sm = p0 + p1;
      sm += __shfl_xor(sm, 1, 32);
      sm += __shfl_xor(sm, 2, 32);
      sm += __shfl_xor(sm, 4, 32);
      sm += __shfl_xor(sm, 8, 32);
      rl[r] = rl[r] * al[r] + sm;
      const int row = r + (half << 3);
      Pw[row * 32 + nn]      = f2bf(p0);
      Pw[row * 32 + 16 + nn] = f2bf(p1);
    }
#pragma unroll
    for (int ch = 0; ch < 32; ++ch)
#pragma unroll
      for (int r = 0; r < 8; ++r) o[ch].f[r] *= al[r];

    // ---- O += P @ V (wave-private P via LDS relayout) ----
    AF pf;
    load_afrag(pf, Pw, 0, 32, 0, lane);
#pragma unroll
    for (int ch = 0; ch < 32; ++ch) {
      AF vf;
      load_bfrag(vf, Vs, ch * 16, 32, 0, lane);
      o[ch].v = wmma_bf16(pf, vf, o[ch].v);
    }
    __syncthreads();   // buffer consumed: safe for next DMA / copy
  }

  float inv[8];
#pragma unroll
  for (int r = 0; r < 8; ++r) inv[r] = 1.0f / rl[r];

  uint16_t* Og = Aout + ((size_t)b * T_ + q0) * NH_;
#pragma unroll
  for (int ch = 0; ch < 32; ++ch)
#pragma unroll
    for (int r = 0; r < 8; ++r) {
      const int row = r + (half << 3);
      Og[(size_t)row * NH_ + ch * 16 + nn] = f2bf(o[ch].f[r] * inv[r]);
    }
}

// ---------------------------------------------------------------------------
// Kernel 3: output projection.  attn(bf16)[16384,512] @ W_o(fp32) -> fp32 out.
// ---------------------------------------------------------------------------
__global__ __launch_bounds__(256) void out_gemm(
    const uint16_t* __restrict__ A, const float* __restrict__ W,
    float* __restrict__ Out)
{
  __shared__ uint16_t As[128 * 32];
  __shared__ uint16_t Bs[128 * 32];   // transposed [n][k]

  const int tid = threadIdx.x, lane = tid & 31, wid = tid >> 5;
  const int row0 = blockIdx.x * 128, n0 = blockIdx.y * 128;
  const int wrow = (wid >> 2) * 64, wcol = (wid & 3) * 32;

  CF acc[4][2];
#pragma unroll
  for (int mt = 0; mt < 4; ++mt)
#pragma unroll
    for (int nt = 0; nt < 2; ++nt)
#pragma unroll
      for (int r = 0; r < 8; ++r) acc[mt][nt].f[r] = 0.f;

  for (int k0 = 0; k0 < NH_; k0 += 32) {
    uint32_t* AsU = (uint32_t*)As;
    const uint32_t* Au = (const uint32_t*)A;
#pragma unroll
    for (int j = 0; j < 8; ++j) {            // 128x32 bf16 copy
      const int e = tid + j * 256;
      const int r = e >> 4, kd = e & 15;
      AsU[e] = Au[(size_t)(row0 + r) * (NH_ / 2) + (k0 >> 1) + kd];
    }
#pragma unroll
    for (int j = 0; j < 4; ++j) {            // 32x128 fp32 -> bf16, transposed
      const int e = tid + j * 256;
      const int kr = e >> 5, nc = (e & 31) << 2;
      const float4 f = *(const float4*)(W + (size_t)(k0 + kr) * NH_ + n0 + nc);
      Bs[(nc + 0) * 32 + kr] = f2bf(f.x);
      Bs[(nc + 1) * 32 + kr] = f2bf(f.y);
      Bs[(nc + 2) * 32 + kr] = f2bf(f.z);
      Bs[(nc + 3) * 32 + kr] = f2bf(f.w);
    }
    __syncthreads();

    AF a[4], b[2];
#pragma unroll
    for (int mt = 0; mt < 4; ++mt) load_afrag(a[mt], As, wrow + mt * 16, 32, 0, lane);
#pragma unroll
    for (int nt = 0; nt < 2; ++nt) load_bfrag(b[nt], Bs, wcol + nt * 16, 32, 0, lane);
#pragma unroll
    for (int mt = 0; mt < 4; ++mt)
#pragma unroll
      for (int nt = 0; nt < 2; ++nt)
        acc[mt][nt].v = wmma_bf16(a[mt], b[nt], acc[mt][nt].v);
    __syncthreads();
  }

  const int half = lane >> 4, nn = lane & 15;
#pragma unroll
  for (int mt = 0; mt < 4; ++mt)
#pragma unroll
    for (int nt = 0; nt < 2; ++nt)
#pragma unroll
      for (int r = 0; r < 8; ++r) {
        const int gm = row0 + wrow + mt * 16 + r + (half << 3);
        const int gn = n0 + wcol + nt * 16 + nn;
        Out[(size_t)gm * NH_ + gn] = acc[mt][nt].f[r];
      }
}

// ---------------------------------------------------------------------------
extern "C" void kernel_launch(void* const* d_in, const int* in_sizes, int n_in,
                              void* d_out, int out_size, void* d_ws, size_t ws_size,
                              hipStream_t stream)
{
  (void)in_sizes; (void)n_in; (void)out_size; (void)ws_size;
  const float* x    = (const float*)d_in[0];
  const int*   mask = (const int*)  d_in[1];
  const float* Wq   = (const float*)d_in[2];
  const float* Wk   = (const float*)d_in[3];
  const float* Wv   = (const float*)d_in[4];
  const float* Wo   = (const float*)d_in[5];
  float* out = (float*)d_out;

  const size_t elems = (size_t)B_ * T_ * NH_;      // 8,388,608
  uint16_t* Qbf = (uint16_t*)d_ws;                 // [b*T][feat]
  uint16_t* Kbf = Qbf + elems;                     // [b*T][feat]
  uint16_t* Vtg = Kbf + elems;                     // transposed [b][feat][T]
  uint16_t* Abf = Vtg + elems;                     // [b*T][feat]

  qkv_gemm<<<dim3((B_ * T_) / 128, NH_ / 128, 3), 256, 0, stream>>>(
      x, Wq, Wk, Wv, Qbf, Kbf, Vtg);

  // double-buffered K/V tiles + per-wave P scratch
  const size_t smem = (4 * 16384 + 8 * 16 * 32) * sizeof(uint16_t); // 139264 B
  flash_attn<<<dim3(T_ / 128, B_), 256, smem, stream>>>(
      Qbf, Kbf, Vtg, mask, Abf);

  out_gemm<<<dim3((B_ * T_) / 128, NH_ / 128), 256, 0, stream>>>(Abf, Wo, out);
}
